// TPA_LSTM_67774583931156
// MI455X (gfx1250) — matrix operand order repaired and checked by reading, not compiled, not run on hardware
//
#include <hip/hip_runtime.h>
#include <hip/hip_bf16.h>
#include <cstddef>
#include <cstdint>

// ---------------------------------------------------------------- constants
#define BB   256
#define TT   256
#define FIN  512
#define EE   512
#define FN   32
#define NL   2
#define HOUT 24
#define ALPHA 0.3f

// ---------------------------------------------------------------- types
typedef __attribute__((ext_vector_type(16))) __bf16 v16bf;
typedef __attribute__((ext_vector_type(8)))  __bf16 v8bf;
typedef __attribute__((ext_vector_type(8)))  float  v8f;
typedef __attribute__((ext_vector_type(4)))  float  vf4;

union FragBF {
    v16bf v;
    struct { v8bf lo; v8bf hi; } h;
};

__device__ __forceinline__ v8f wmma_bf16(v16bf a, v16bf b, v8f c) {
    // D = A(16x32 bf16) * B(32x16 bf16) + C(16x16 f32)
    return __builtin_amdgcn_wmma_f32_16x16x32_bf16(
        false, a, false, b, (short)0, c, false, false);
}

// ---- gfx1250 async copy: 16B global -> LDS, tracked by ASYNCcnt ----------
// ISA 15.18.3 opcode 98: GLOBAL_LOAD_ASYNC_TO_LDS_B128 (VDST = LDS addr VGPR).
__device__ __forceinline__ void async_cp16(const void* gptr, void* lptr) {
    unsigned lds = (unsigned)(unsigned long long)lptr;  // low 32 bits give the LDS address
    asm volatile("global_load_async_to_lds_b128 %0, %1, off"
                 :: "v"(lds), "v"(gptr) : "memory");
}
__device__ __forceinline__ void wait_async0() {
    asm volatile("s_wait_asynccnt 0x0" ::: "memory");
}

// A fragment (16x32 bf16), A row-major with leading dim lda.
// lane L (<16): row = m0+L,   K-chunks {k0..k0+7, k0+16..k0+23}
// lane L (>=16): row = m0+L-16, K-chunks {k0+8..k0+15, k0+24..k0+31}
__device__ __forceinline__ v16bf loadA_bf16(const __bf16* A, int lda,
                                            int m0, int k0, int lane) {
    const int row = m0 + (lane & 15);
    const int kk  = k0 + ((lane >> 4) << 3);
    const __bf16* p = A + (size_t)row * lda + kk;
    FragBF f;
    f.h.lo = *(const v8bf*)(p);
    f.h.hi = *(const v8bf*)(p + 16);
    return f.v;
}

// Same fragment but source is f32 (converted to bf16 in registers).
__device__ __forceinline__ v16bf loadA_f32(const float* A, int lda,
                                           int m0, int k0, int lane) {
    const int row = m0 + (lane & 15);
    const int kk  = k0 + ((lane >> 4) << 3);
    const float* p = A + (size_t)row * lda + kk;
    vf4 q0 = *(const vf4*)(p);
    vf4 q1 = *(const vf4*)(p + 4);
    vf4 q2 = *(const vf4*)(p + 16);
    vf4 q3 = *(const vf4*)(p + 20);
    v16bf a;
#pragma unroll
    for (int i = 0; i < 4; ++i) {
        a[i]      = (__bf16)q0[i];
        a[4 + i]  = (__bf16)q1[i];
        a[8 + i]  = (__bf16)q2[i];
        a[12 + i] = (__bf16)q3[i];
    }
    return a;
}

// B fragment (32x16 bf16) read from an LDS-resident row-major tile.
// lane L holds column n0+L%16, K = {8h..8h+7, 16+8h..16+8h+7}, h = L/16
__device__ __forceinline__ v16bf loadB_lds(const __bf16* base, int ldb,
                                           int n0, int lane) {
    const __bf16* p = base + n0 + (lane & 15);
    const int kb = (lane >> 4) << 3;
    v16bf b;
#pragma unroll
    for (int i = 0; i < 8; ++i) {
        b[i]     = p[(size_t)(kb + i) * ldb];
        b[8 + i] = p[(size_t)(kb + 16 + i) * ldb];
    }
    return b;
}

__device__ __forceinline__ float sigm(float x) { return 1.f / (1.f + __expf(-x)); }
__device__ __forceinline__ float lky(float v)  { return v >= 0.f ? v : ALPHA * v; }

// ---------------------------------------------------------------- cvt kernels
__global__ void k_cvt(const float* __restrict__ src, __bf16* __restrict__ dst, int n) {
    int i = blockIdx.x * blockDim.x + threadIdx.x;
    if (i < n) dst[i] = (__bf16)src[i];
}

// conv_K (255,32) f32 -> (256,32) bf16, zero-padded row 255
__global__ void k_cvt_pad_convK(const float* __restrict__ src, __bf16* __restrict__ dst) {
    int i = blockIdx.x * blockDim.x + threadIdx.x;
    if (i < 256 * FN) {
        int l = i >> 5, f = i & 31;
        dst[i] = (l < TT - 1) ? (__bf16)src[l * FN + f] : (__bf16)0.f;
    }
}

// ---------------------------------------------------------------- GEMM1: embedding
// e = leaky(x @ W_emb + b_emb), written TRANSPOSED as h0 in (B,E,T) bf16 layout.
// grid = (BT/256, E/16); block = 256 (8 waves stacked along M).
// B tile (32x16) is shared by all 8 waves -> staged via async copy into LDS.
__global__ __launch_bounds__(256) void k_emb_gemm(
    const float* __restrict__ x,      // (B*T, FIN) f32
    const __bf16* __restrict__ Wb,    // (FIN, E)  bf16
    const float* __restrict__ bias,   // (E)
    __bf16* __restrict__ h0)          // (B*E, T)  bf16
{
    __shared__ __bf16 Bs[2][32][16];                 // 2 KB double buffer
    const int tid  = threadIdx.x;
    const int lane = tid & 31;
    const int wave = tid >> 5;
    const int m0 = blockIdx.x * 256 + wave * 32;
    const int n0 = blockIdx.y * 16;
    const int NS = FIN / 32;

    // stage K-step 0
    if (tid < 64) {
        int r = tid >> 1, half = tid & 1;
        async_cp16(Wb + (size_t)r * EE + n0 + half * 8, &Bs[0][r][half * 8]);
    }

    v8f c0 = {}, c1 = {};
    for (int s = 0; s < NS; ++s) {
        const int k0  = s * 32;
        const int cur = s & 1;
        wait_async0();
        __syncthreads();                             // copies into `cur` visible
        if (s + 1 < NS && tid < 64) {                // prefetch next into cur^1
            int r = tid >> 1, half = tid & 1;
            async_cp16(Wb + (size_t)(k0 + 32 + r) * EE + n0 + half * 8,
                       &Bs[cur ^ 1][r][half * 8]);
        }
        if (k0 + 32 < FIN)
            __builtin_prefetch(x + (size_t)(m0 + (lane & 15)) * FIN + k0 + 32, 0, 3);
        v16bf a0 = loadA_f32(x, FIN, m0,      k0, lane);
        v16bf a1 = loadA_f32(x, FIN, m0 + 16, k0, lane);
        v16bf b  = loadB_lds(&Bs[cur][0][0], 16, 0, lane);
        c0 = wmma_bf16(a0, b, c0);
        c1 = wmma_bf16(a1, b, c1);
        __syncthreads();                             // done reading before overwrite
    }

    const int n     = n0 + (lane & 15);
    const float bn  = bias[n];
    const int rbase = (lane >> 4) << 3;   // 0 or 8
#pragma unroll
    for (int sub = 0; sub < 2; ++sub) {
        v8f c = sub ? c1 : c0;
        const int m   = m0 + sub * 16 + rbase;  // 8 consecutive rows m..m+7
        const int bi  = m >> 8;                 // / T
        const int t   = m & 255;                // % T
        __bf16* dst = h0 + ((size_t)(bi * EE + n)) * TT + t;
        v8bf out;
#pragma unroll
        for (int r = 0; r < 8; ++r) out[r] = (__bf16)lky(c[r] + bn);
        *(v8bf*)dst = out;                      // 8 consecutive t -> 16B store
    }
}

// ---------------------------------------------------------------- LSTM layer
// z = h @ W + b ; gates i/g/o computed in-register (f gate unused in reference).
// h_new = sigmoid(o) * tanh( sigmoid(i) * tanh(g) )
// grid = (B*E/256, T/16); block = 256.  3 gate B-tiles staged async into LDS.
template <bool FINAL>
__global__ __launch_bounds__(256) void k_lstm(
    const __bf16* __restrict__ hin,   // (B*E, T) bf16
    const __bf16* __restrict__ W,     // (T, 4T)  bf16
    const float* __restrict__ bvec,   // (4T)
    __bf16* __restrict__ hout,        // (B*E, T) bf16
    float* __restrict__ ht)           // (B*E) f32, only if FINAL
{
    __shared__ __bf16 Bs[2][3][32][16];              // 6 KB double buffer
    const int tid  = threadIdx.x;
    const int lane = tid & 31;
    const int wave = tid >> 5;
    const int m0 = blockIdx.x * 256 + wave * 32;
    const int t0 = blockIdx.y * 16;
    const int NS = TT / 32;
    // gate column bases in W (f gate skipped)
    const int gcol0 = 0 * TT + t0, gcol1 = 2 * TT + t0, gcol2 = 3 * TT + t0;

    if (tid < 192) {
        int gate = tid >> 6, rem = tid & 63, r = rem >> 1, half = rem & 1;
        int gc = gate == 0 ? gcol0 : (gate == 1 ? gcol1 : gcol2);
        async_cp16(W + (size_t)r * (4 * TT) + gc + half * 8, &Bs[0][gate][r][half * 8]);
    }

    v8f ai[2] = {{}, {}};
    v8f ag[2] = {{}, {}};
    v8f ao[2] = {{}, {}};

    for (int s = 0; s < NS; ++s) {
        const int k0  = s * 32;
        const int cur = s & 1;
        wait_async0();
        __syncthreads();
        if (s + 1 < NS && tid < 192) {
            int gate = tid >> 6, rem = tid & 63, r = rem >> 1, half = rem & 1;
            int gc = gate == 0 ? gcol0 : (gate == 1 ? gcol1 : gcol2);
            async_cp16(W + (size_t)(k0 + 32 + r) * (4 * TT) + gc + half * 8,
                       &Bs[cur ^ 1][gate][r][half * 8]);
        }
        if (k0 + 32 < TT)
            __builtin_prefetch(hin + (size_t)(m0 + (lane & 15)) * TT + k0 + 32, 0, 3);
        v16bf a0 = loadA_bf16(hin, TT, m0,      k0, lane);
        v16bf a1 = loadA_bf16(hin, TT, m0 + 16, k0, lane);
        v16bf bi = loadB_lds(&Bs[cur][0][0][0], 16, 0, lane);
        v16bf bg = loadB_lds(&Bs[cur][1][0][0], 16, 0, lane);
        v16bf bo = loadB_lds(&Bs[cur][2][0][0], 16, 0, lane);
        ai[0] = wmma_bf16(a0, bi, ai[0]);  ai[1] = wmma_bf16(a1, bi, ai[1]);
        ag[0] = wmma_bf16(a0, bg, ag[0]);  ag[1] = wmma_bf16(a1, bg, ag[1]);
        ao[0] = wmma_bf16(a0, bo, ao[0]);  ao[1] = wmma_bf16(a1, bo, ao[1]);
        __syncthreads();
    }

    const int t = t0 + (lane & 15);
    const float bi_ = bvec[0 * TT + t];
    const float bg_ = bvec[2 * TT + t];
    const float bo_ = bvec[3 * TT + t];
    const int rbase = (lane >> 4) << 3;
#pragma unroll
    for (int sub = 0; sub < 2; ++sub) {
#pragma unroll
        for (int r = 0; r < 8; ++r) {
            const int m = m0 + sub * 16 + rbase + r;
            float ig = sigm(ai[sub][r] + bi_);
            float gg = tanhf(ag[sub][r] + bg_);
            float og = sigm(ao[sub][r] + bo_);
            float hv = og * tanhf(ig * gg);
            if (FINAL) {
                hv = lky(hv);
                hout[(size_t)m * TT + t] = (__bf16)hv;
                if (t == TT - 1) ht[m] = hv;
            } else {
                hout[(size_t)m * TT + t] = (__bf16)hv;
            }
        }
    }
}

// ---------------------------------------------------------------- conv einsum GEMM
// H[b,e,f] = sum_l hm[b,e,l] * K[l,f] + cb[f]; K zero-padded to 256 rows.
// grid = (B*E/256, 1); block = 256. Each wave: 32 rows x 32 cols.
__global__ __launch_bounds__(256) void k_conv(
    const __bf16* __restrict__ hm,   // (B*E, T) bf16 (leaky applied)
    const __bf16* __restrict__ Kb,   // (256, FN) bf16 padded
    const float* __restrict__ cb,    // (FN)
    float* __restrict__ H)           // (B*E, FN) f32
{
    __shared__ __bf16 Bs[2][32][32];                 // 4 KB double buffer
    const int tid  = threadIdx.x;
    const int lane = tid & 31;
    const int wave = tid >> 5;
    const int m0 = blockIdx.x * 256 + wave * 32;
    const int NS = TT / 32;

    if (tid < 128) {
        int r = tid >> 2, q = tid & 3;
        async_cp16(Kb + (size_t)r * FN + q * 8, &Bs[0][r][q * 8]);
    }

    v8f c00 = {}, c01 = {}, c10 = {}, c11 = {};
    for (int s = 0; s < NS; ++s) {
        const int k0  = s * 32;
        const int cur = s & 1;
        wait_async0();
        __syncthreads();
        if (s + 1 < NS && tid < 128) {
            int r = tid >> 2, q = tid & 3;
            async_cp16(Kb + (size_t)(k0 + 32 + r) * FN + q * 8, &Bs[cur ^ 1][r][q * 8]);
        }
        v16bf a0 = loadA_bf16(hm, TT, m0,      k0, lane);
        v16bf a1 = loadA_bf16(hm, TT, m0 + 16, k0, lane);
        v16bf b0 = loadB_lds(&Bs[cur][0][0], 32, 0,  lane);
        v16bf b1 = loadB_lds(&Bs[cur][0][0], 32, 16, lane);
        c00 = wmma_bf16(a0, b0, c00);  c01 = wmma_bf16(a0, b1, c01);
        c10 = wmma_bf16(a1, b0, c10);  c11 = wmma_bf16(a1, b1, c11);
        __syncthreads();
    }

    const int nl    = lane & 15;
    const int rbase = (lane >> 4) << 3;
#pragma unroll
    for (int msub = 0; msub < 2; ++msub) {
#pragma unroll
        for (int nsub = 0; nsub < 2; ++nsub) {
            v8f c = msub ? (nsub ? c11 : c10) : (nsub ? c01 : c00);
            const int n = nsub * 16 + nl;
            const float bn = cb[n];
#pragma unroll
            for (int r = 0; r < 8; ++r) {
                const int m = m0 + msub * 16 + rbase + r;
                H[(size_t)m * FN + n] = c[r] + bn;
            }
        }
    }
}

// ---------------------------------------------------------------- small tail ops
__global__ void k_htscore(const float* __restrict__ ht, const float* __restrict__ Ww,
                          const float* __restrict__ bw, float* __restrict__ hts) {
    int i = blockIdx.x * blockDim.x + threadIdx.x;   // B*FN
    if (i >= BB * FN) return;
    int b = i >> 5, f = i & 31;
    float acc = bw[f];
    const float* hr = ht + (size_t)b * EE;
    for (int k = 0; k < EE; ++k) acc += hr[k] * Ww[k * FN + f];
    hts[i] = acc;
}

__global__ __launch_bounds__(256) void k_attn(
    const float* __restrict__ H,     // (B*E, FN)
    const float* __restrict__ hts,   // (B, FN)
    float* __restrict__ vt)          // (B, FN)
{
    __shared__ float s[FN];
    __shared__ float a[EE];
    __shared__ float part[8][FN];
    const int b = blockIdx.x, tid = threadIdx.x;

    if (tid < FN) s[tid] = hts[b * FN + tid];
    __syncthreads();

    for (int e = tid; e < EE; e += 256) {
        const float* Hr = H + ((size_t)(b * EE + e)) * FN;
        float sc = 0.f;
#pragma unroll
        for (int f = 0; f < FN; ++f) sc += Hr[f] * s[f];
        a[e] = 1.f / (1.f + __expf(-sc));
    }
    __syncthreads();

    const int f = tid & 31, g = tid >> 5;  // 32 cols x 8 groups
    float p = 0.f;
    for (int e = g * 64; e < g * 64 + 64; ++e)
        p += a[e] * H[((size_t)(b * EE + e)) * FN + f];
    part[g][f] = p;
    __syncthreads();

    if (tid < FN) {
        float acc = 0.f;
#pragma unroll
        for (int gg = 0; gg < 8; ++gg) acc += part[gg][tid];
        vt[b * FN + tid] = acc;
    }
}

__global__ void k_fc1(const float* __restrict__ vt, const float* __restrict__ ht,
                      const float* __restrict__ Wf1, const float* __restrict__ bf1,
                      float* __restrict__ mid) {
    int i = blockIdx.x * blockDim.x + threadIdx.x;   // B*E
    if (i >= BB * EE) return;
    int b = i >> 9, j = i & 511;
    float acc = bf1[j];
    for (int k = 0; k < FN; ++k) acc += vt[b * FN + k] * Wf1[k * EE + j];
    const float* hr = ht + (size_t)b * EE;
    for (int k = 0; k < EE; ++k) acc += hr[k] * Wf1[(FN + k) * EE + j];
    mid[i] = acc;
}

__global__ void k_fc2(const float* __restrict__ mid, const float* __restrict__ Wf2,
                      const float* __restrict__ bf2, float* __restrict__ out) {
    int i = blockIdx.x * blockDim.x + threadIdx.x;   // B*HOUT
    if (i >= BB * HOUT) return;
    int b = i / HOUT, o = i % HOUT;
    float acc = bf2[o];
    const float* mr = mid + (size_t)b * EE;
    for (int k = 0; k < EE; ++k) acc += mr[k] * Wf2[k * HOUT + o];
    out[i] = acc;
}

// ---------------------------------------------------------------- launch
extern "C" void kernel_launch(void* const* d_in, const int* in_sizes, int n_in,
                              void* d_out, int out_size, void* d_ws, size_t ws_size,
                              hipStream_t stream) {
    const float* x      = (const float*)d_in[0];   // (B,T,FIN)
    const float* W_emb  = (const float*)d_in[1];   // (FIN,E)
    const float* b_emb  = (const float*)d_in[2];   // (E)
    const float* lstm_W = (const float*)d_in[3];   // (NL,T,4T)
    const float* lstm_b = (const float*)d_in[4];   // (NL,4T)
    const float* conv_K = (const float*)d_in[5];   // (T-1,FN)
    const float* conv_b = (const float*)d_in[6];   // (FN)
    const float* Ww     = (const float*)d_in[7];   // (E,FN)
    const float* bw     = (const float*)d_in[8];   // (FN)
    const float* Wf1    = (const float*)d_in[9];   // (FN+E,E)
    const float* bf1    = (const float*)d_in[10];  // (E)
    const float* Wf2    = (const float*)d_in[11];  // (E,HOUT)
    const float* bf2    = (const float*)d_in[12];  // (HOUT)
    float* out = (float*)d_out;

    char* ws = (char*)d_ws;
    const size_t SZ_H = (size_t)BB * EE * TT * sizeof(__bf16);     // 67,108,864
    const size_t OFF_H1    = 0;
    const size_t OFF_H0    = OFF_H1 + SZ_H;       // h0, later reused for h2m
    const size_t OFF_HBUF  = OFF_H0 + SZ_H;       // H f32: B*E*FN*4
    const size_t OFF_WEMB  = OFF_HBUF + (size_t)BB * EE * FN * 4;
    const size_t OFF_LSTMW = OFF_WEMB + (size_t)FIN * EE * 2;
    const size_t OFF_CONVK = OFF_LSTMW + (size_t)NL * TT * 4 * TT * 2;
    const size_t OFF_HT    = OFF_CONVK + (size_t)256 * FN * 2;
    const size_t OFF_HTS   = OFF_HT + (size_t)BB * EE * 4;
    const size_t OFF_VT    = OFF_HTS + (size_t)BB * FN * 4;
    const size_t OFF_MID   = OFF_VT + (size_t)BB * FN * 4;

    __bf16* h1      = (__bf16*)(ws + OFF_H1);
    __bf16* h0      = (__bf16*)(ws + OFF_H0);
    float*  Hbuf    = (float*)(ws + OFF_HBUF);
    __bf16* WembB   = (__bf16*)(ws + OFF_WEMB);
    __bf16* lstmWB  = (__bf16*)(ws + OFF_LSTMW);
    __bf16* convKB  = (__bf16*)(ws + OFF_CONVK);
    float*  ht      = (float*)(ws + OFF_HT);
    float*  hts     = (float*)(ws + OFF_HTS);
    float*  vt      = (float*)(ws + OFF_VT);
    float*  mid     = (float*)(ws + OFF_MID);

    // weight conversions to bf16
    k_cvt<<<(FIN * EE + 255) / 256, 256, 0, stream>>>(W_emb, WembB, FIN * EE);
    k_cvt<<<(NL * TT * 4 * TT + 255) / 256, 256, 0, stream>>>(lstm_W, lstmWB, NL * TT * 4 * TT);
    k_cvt_pad_convK<<<(256 * FN + 255) / 256, 256, 0, stream>>>(conv_K, convKB);

    // GEMM1: embedding + leaky + transpose into (B,E,T)
    k_emb_gemm<<<dim3(BB * TT / 256, EE / 16), 256, 0, stream>>>(x, WembB, b_emb, h0);

    // LSTM layers (f gate skipped)
    k_lstm<false><<<dim3(BB * EE / 256, TT / 16), 256, 0, stream>>>(
        h0, lstmWB, lstm_b, h1, (float*)nullptr);
    k_lstm<true><<<dim3(BB * EE / 256, TT / 16), 256, 0, stream>>>(
        h1, lstmWB + (size_t)TT * 4 * TT, lstm_b + 4 * TT, h0 /* h2m reuses h0 */, ht);

    // conv einsum GEMM
    k_conv<<<dim3(BB * EE / 256, 1), 256, 0, stream>>>(h0, convKB, conv_b, Hbuf);

    // attention + FC head
    k_htscore<<<(BB * FN + 255) / 256, 256, 0, stream>>>(ht, Ww, bw, hts);
    k_attn<<<BB, 256, 0, stream>>>(Hbuf, hts, vt);
    k_fc1<<<(BB * EE + 255) / 256, 256, 0, stream>>>(vt, ht, Wf1, bf1, mid);
    k_fc2<<<(BB * HOUT + 255) / 256, 256, 0, stream>>>(mid, Wf2, bf2, out);
}
